// Qwen3Attention_23493471109627
// MI455X (gfx1250) — matrix operand order repaired and checked by reading, not compiled, not run on hardware
//
#include <hip/hip_runtime.h>

// ---------------------------------------------------------------------------
// Qwen3 attention block for MI455X (gfx1250, wave32, WMMA bf16)
// B=64 S=128 HID=4096 H=32 HKV=8 D=128 G=4
// ---------------------------------------------------------------------------

typedef __bf16 bf16x16 __attribute__((ext_vector_type(16)));
typedef float  f32x8   __attribute__((ext_vector_type(8)));

union FragU {
  bf16x16 v;
  unsigned int u[8];
  uint4 q2[2];
};

__device__ __forceinline__ unsigned short f32_to_bf16(float f) {
  unsigned int u = __float_as_uint(f);
  u += 0x7FFFu + ((u >> 16) & 1u);   // round-to-nearest-even
  return (unsigned short)(u >> 16);
}

constexpr int   kB    = 64;
constexpr int   kS    = 128;
constexpr int   kH    = 32;
constexpr int   kHKV  = 8;
constexpr int   kD    = 128;
constexpr int   kHID  = 4096;
constexpr float kScale = 0.08838834764831845f;  // 1/sqrt(128)
constexpr float kNeg   = -10000.0f;
constexpr int   LSTR   = 40;   // LDS row stride (bf16 elems): 80 B, 16B-aligned rows
constexpr int   PSTR   = 136;  // attention P row stride: 272 B, 16B-aligned rows

// ---------------------------------------------------------------------------
// GEMM: C[M,N] = A[M,K] * B[N,K]^T   (fp32 accumulate, bf16 WMMA inputs)
// A either fp32 (converted while staging to LDS) or already bf16.
// 128x128 block tile, 8 waves (4M x 2N), wave = 32x64 (2x4 WMMA tiles),
// double-buffered LDS: one barrier per K-step, global loads overlap WMMAs.
// ---------------------------------------------------------------------------
template <bool A_BF16>
__device__ __forceinline__ void stage_issue(
    const void* __restrict__ Ap, const float* __restrict__ Bp,
    int m0, int n0, int K, int kk, int t,
    float* fa, unsigned short* ha, float* fb) {
#pragma unroll
  for (int i = 0; i < 16; ++i) {
    int e = i * 256 + t;
    int row = e >> 5, k = e & 31;
    if (A_BF16)
      ha[i] = ((const unsigned short*)Ap)[(size_t)(m0 + row) * K + kk + k];
    else
      fa[i] = ((const float*)Ap)[(size_t)(m0 + row) * K + kk + k];
    fb[i] = Bp[(size_t)(n0 + row) * K + kk + k];
  }
}

template <bool A_BF16>
__device__ __forceinline__ void stage_commit(
    unsigned short* __restrict__ Ab, unsigned short* __restrict__ Bb, int t,
    const float* fa, const unsigned short* ha, const float* fb) {
#pragma unroll
  for (int i = 0; i < 16; ++i) {
    int e = i * 256 + t;
    int row = e >> 5, k = e & 31;
    Ab[row * LSTR + k] = A_BF16 ? ha[i] : f32_to_bf16(fa[i]);
    Bb[row * LSTR + k] = f32_to_bf16(fb[i]);
  }
}

template <bool A_BF16>
__global__ __launch_bounds__(256) void gemm_xWT(
    const void* __restrict__ Ap, const float* __restrict__ Bp,
    float* __restrict__ Cp, int M, int N, int K) {
  __shared__ __align__(16) unsigned short Abuf[2][128 * LSTR];
  __shared__ __align__(16) unsigned short Bbuf[2][128 * LSTR];
  (void)M;
  const int t    = threadIdx.x;
  const int m0   = blockIdx.y * 128;
  const int n0   = blockIdx.x * 128;
  const int wid  = t >> 5, lane = t & 31;
  const int waveM = wid & 3, waveN = wid >> 2;
  const int half  = lane >> 4, l16 = lane & 15;
  const int nk = K >> 5;

  f32x8 acc[2][4];
  for (int i = 0; i < 2; ++i)
    for (int j = 0; j < 4; ++j)
      for (int e = 0; e < 8; ++e) acc[i][j][e] = 0.0f;

  float fa[16]; unsigned short ha[16]; float fb[16];

  stage_issue<A_BF16>(Ap, Bp, m0, n0, K, 0, t, fa, ha, fb);
  stage_commit<A_BF16>(Abuf[0], Bbuf[0], t, fa, ha, fb);
  __syncthreads();

  for (int kt = 0; kt < nk; ++kt) {
    const int cur = kt & 1;
    if (kt + 1 < nk)  // prefetch next tile into registers; hides behind WMMAs
      stage_issue<A_BF16>(Ap, Bp, m0, n0, K, (kt + 1) * 32, t, fa, ha, fb);

    const unsigned short* Ab = Abuf[cur];
    const unsigned short* Bb = Bbuf[cur];
    FragU a[2], b[4];
#pragma unroll
    for (int mt = 0; mt < 2; ++mt) {
      // A 16x32 bf16 layout: lane-half K ranges {8h..8h+7} and {16+8h..16+8h+7},
      // each a contiguous 16-byte run -> two ds_load_b128.
      int base = (waveM * 32 + mt * 16 + l16) * LSTR;
      a[mt].q2[0] = *(const uint4*)&Ab[base + 8 * half];
      a[mt].q2[1] = *(const uint4*)&Ab[base + 16 + 8 * half];
    }
#pragma unroll
    for (int nt = 0; nt < 4; ++nt) {
      // B 32x16 bf16 layout: lane-half K ranges {8h..8h+7} and {16+8h..16+8h+7} -> two ds_load_b128.
      int base = (waveN * 64 + nt * 16 + l16) * LSTR;
      b[nt].q2[0] = *(const uint4*)&Bb[base + 8 * half];
      b[nt].q2[1] = *(const uint4*)&Bb[base + 16 + 8 * half];
    }
#pragma unroll
    for (int mt = 0; mt < 2; ++mt)
#pragma unroll
      for (int nt = 0; nt < 4; ++nt)
        acc[mt][nt] = __builtin_amdgcn_wmma_f32_16x16x32_bf16(
            false, a[mt].v, false, b[nt].v, (short)0, acc[mt][nt], false, false);

    if (kt + 1 < nk)
      stage_commit<A_BF16>(Abuf[1 - cur], Bbuf[1 - cur], t, fa, ha, fb);
    __syncthreads();
  }

#pragma unroll
  for (int mt = 0; mt < 2; ++mt)
    for (int nt = 0; nt < 4; ++nt)
      for (int r = 0; r < 8; ++r) {
        int grow = m0 + waveM * 32 + mt * 16 + half * 8 + r;
        int gcol = n0 + waveN * 64 + nt * 16 + l16;
        Cp[(size_t)grow * N + gcol] = acc[mt][nt][r];
      }
}

// ---------------------------------------------------------------------------
// Per-head RMSNorm (fp32) + RoPE, fp32 [B*S, nheads*D] -> bf16 [B, nheads, S, D]
// ---------------------------------------------------------------------------
__global__ __launch_bounds__(128) void rmsnorm_rope(
    const float* __restrict__ x, const float* __restrict__ w,
    const float* __restrict__ cosT, const float* __restrict__ sinT,
    unsigned short* __restrict__ out, int nheads) {
  __shared__ float red[128];
  __shared__ float nrm[128];
  int idx = blockIdx.x;
  int h = idx % nheads;
  int s = (idx / nheads) % kS;
  int b = idx / (nheads * kS);
  int d = threadIdx.x;
  float xv = x[(((size_t)(b * kS + s)) * nheads + h) * kD + d];
  red[d] = xv * xv;
  __syncthreads();
  for (int off = 64; off > 0; off >>= 1) {
    if (d < off) red[d] += red[d + off];
    __syncthreads();
  }
  float inv = rsqrtf(red[0] * (1.0f / kD) + 1e-6f);
  float nv = xv * inv * w[d];
  nrm[d] = nv;
  __syncthreads();
  float c = cosT[s * kD + d], sn = sinT[s * kD + d];
  float rot = (d < 64) ? -nrm[d + 64] : nrm[d - 64];
  out[((((size_t)b * nheads + h) * kS + s)) * kD + d] = f32_to_bf16(nv * c + rot * sn);
}

// V: fp32 [B*S, HKV*D] -> bf16 [B, HKV, S, D]
__global__ __launch_bounds__(256) void v_permute(
    const float* __restrict__ v, unsigned short* __restrict__ out) {
  int tid = blockIdx.x * 256 + threadIdx.x;
  int d  = tid & 127;
  int kv = (tid >> 7) & 7;
  int s  = (tid >> 10) & 127;
  int b  = tid >> 17;
  out[((((size_t)b * kHKV + kv) * kS + s)) * kD + d] =
      f32_to_bf16(v[(((size_t)(b * kS + s)) * kHKV + kv) * kD + d]);
}

// ---------------------------------------------------------------------------
// Causal GQA attention for one (b, h): S=128, D=128.
// Scores via WMMA (Q/K fragments straight from global, merged to b128 loads);
// softmax in registers (half-wave shfl_xor reductions); P bf16 in LDS;
// P*V via WMMA with V fragments packed from global.
// ---------------------------------------------------------------------------
__global__ __launch_bounds__(256) void attn_kernel(
    const unsigned short* __restrict__ q, const unsigned short* __restrict__ kmat,
    const unsigned short* __restrict__ v, unsigned short* __restrict__ ctx) {
  __shared__ __align__(16) unsigned short Pl[128 * PSTR];  // ~34 KB
  const int t   = threadIdx.x;
  const int b   = blockIdx.x >> 5;
  const int h   = blockIdx.x & 31;
  const int kvh = h >> 2;  // G = 4
  const unsigned short* Qb = q    + (((size_t)b * kH   + h)   * kS) * kD;
  const unsigned short* Kb = kmat + (((size_t)b * kHKV + kvh) * kS) * kD;
  const unsigned short* Vb = v    + (((size_t)b * kHKV + kvh) * kS) * kD;
  const int wid = t >> 5, lane = t & 31;
  const int half = lane >> 4, l16 = lane & 15;
  const int qrow = wid * 16 + l16;

  // ---- phase 1: scores = Q K^T (wave owns q-rows wid*16..+15, all 128 keys)
  f32x8 acc[8];
  for (int nt = 0; nt < 8; ++nt)
    for (int e = 0; e < 8; ++e) acc[nt][e] = 0.0f;
  for (int k0 = 0; k0 < kD; k0 += 32) {
    FragU a;
    {
      const unsigned short* p = &Qb[qrow * kD + k0];
      a.q2[0] = *(const uint4*)&p[8 * half];
      a.q2[1] = *(const uint4*)&p[16 + 8 * half];
    }
#pragma unroll
    for (int nt = 0; nt < 8; ++nt) {
      FragU bfr;
      const unsigned short* p = &Kb[(nt * 16 + l16) * kD + k0];
      bfr.q2[0] = *(const uint4*)&p[8 * half];
      bfr.q2[1] = *(const uint4*)&p[16 + 8 * half];
      acc[nt] = __builtin_amdgcn_wmma_f32_16x16x32_bf16(
          false, a.v, false, bfr.v, (short)0, acc[nt], false, false);
    }
  }

  // ---- scale + causal mask + stable softmax per row, in registers
#pragma unroll
  for (int r = 0; r < 8; ++r) {
    int row = wid * 16 + half * 8 + r;
    float vals[8];
    float mx = -3.4e38f;
#pragma unroll
    for (int nt = 0; nt < 8; ++nt) {
      int col = nt * 16 + l16;
      float sc = acc[nt][r] * kScale + (col > row ? kNeg : 0.0f);
      vals[nt] = sc;
      mx = fmaxf(mx, sc);
    }
    for (int off = 1; off < 16; off <<= 1) mx = fmaxf(mx, __shfl_xor(mx, off, 32));
    float sum = 0.0f;
#pragma unroll
    for (int nt = 0; nt < 8; ++nt) { vals[nt] = __expf(vals[nt] - mx); sum += vals[nt]; }
    for (int off = 1; off < 16; off <<= 1) sum += __shfl_xor(sum, off, 32);
    float is = 1.0f / sum;
#pragma unroll
    for (int nt = 0; nt < 8; ++nt)
      Pl[row * PSTR + nt * 16 + l16] = f32_to_bf16(vals[nt] * is);
  }
  __syncthreads();

  // ---- phase 2: ctx = P V (K = key position)
  f32x8 acc2[8];
  for (int nt = 0; nt < 8; ++nt)
    for (int e = 0; e < 8; ++e) acc2[nt][e] = 0.0f;
  for (int k0 = 0; k0 < kS; k0 += 32) {
    FragU a;
    {
      const unsigned short* p = &Pl[qrow * PSTR + k0];
      a.q2[0] = *(const uint4*)&p[8 * half];
      a.q2[1] = *(const uint4*)&p[16 + 8 * half];
    }
#pragma unroll
    for (int nt = 0; nt < 8; ++nt) {
      FragU bfr;
      int dcol = nt * 16 + l16;
#pragma unroll
      for (int j = 0; j < 8; ++j) {
        int ks = 8 * half + 2 * j + (j & 4) * 2 + k0;
        unsigned int lo = Vb[(size_t)ks * kD + dcol];
        unsigned int hi = Vb[(size_t)(ks + 1) * kD + dcol];
        bfr.u[j] = lo | (hi << 16);
      }
      acc2[nt] = __builtin_amdgcn_wmma_f32_16x16x32_bf16(
          false, a.v, false, bfr.v, (short)0, acc2[nt], false, false);
    }
  }
  // store bf16 ctx as [B, S, H*D] so the O-projection GEMM reads it row-major
#pragma unroll
  for (int nt = 0; nt < 8; ++nt)
    for (int r = 0; r < 8; ++r) {
      int row = wid * 16 + half * 8 + r;
      int d = nt * 16 + l16;
      ctx[((size_t)(b * kS) + row) * kHID + h * kD + d] = f32_to_bf16(acc2[nt][r]);
    }
}

// ---------------------------------------------------------------------------
extern "C" void kernel_launch(void* const* d_in, const int* in_sizes, int n_in,
                              void* d_out, int out_size, void* d_ws, size_t ws_size,
                              hipStream_t stream) {
  (void)in_sizes; (void)n_in; (void)out_size; (void)ws_size;
  const float* hs   = (const float*)d_in[0];
  const float* cosT = (const float*)d_in[1];
  const float* sinT = (const float*)d_in[2];
  const float* wq   = (const float*)d_in[3];
  const float* wk   = (const float*)d_in[4];
  const float* wv   = (const float*)d_in[5];
  const float* wo   = (const float*)d_in[6];
  const float* qnw  = (const float*)d_in[7];
  const float* knw  = (const float*)d_in[8];
  float* out = (float*)d_out;

  // workspace layout (total ~352 MB)
  char* ws = (char*)d_ws;
  float*          q_f32  = (float*)(ws);                        // 128 MB [B*S, H*D]
  float*          k_f32  = (float*)(ws + 134217728);            // 32 MB
  float*          v_f32  = (float*)(ws + 167772160);            // 32 MB
  unsigned short* q_rope = (unsigned short*)(ws + 201326592);   // 64 MB  [B,H,S,D]
  unsigned short* k_rope = (unsigned short*)(ws + 268435456);   // 16 MB  [B,HKV,S,D]
  unsigned short* v_bf   = (unsigned short*)(ws + 285212672);   // 16 MB  [B,HKV,S,D]
  unsigned short* ctx    = (unsigned short*)(ws + 301989888);   // 64 MB  [B,S,H*D]

  const int M = kB * kS;  // 8192
  dim3 blk(256);

  gemm_xWT<false><<<dim3(kHID / 128, M / 128), blk, 0, stream>>>(hs, wq, q_f32, M, kHID, kHID);
  gemm_xWT<false><<<dim3((kHKV * kD) / 128, M / 128), blk, 0, stream>>>(hs, wk, k_f32, M, kHKV * kD, kHID);
  gemm_xWT<false><<<dim3((kHKV * kD) / 128, M / 128), blk, 0, stream>>>(hs, wv, v_f32, M, kHKV * kD, kHID);

  rmsnorm_rope<<<kB * kS * kH,   128, 0, stream>>>(q_f32, qnw, cosT, sinT, q_rope, kH);
  rmsnorm_rope<<<kB * kS * kHKV, 128, 0, stream>>>(k_f32, knw, cosT, sinT, k_rope, kHKV);
  v_permute<<<(kB * kS * kHKV * kD) / 256, 256, 0, stream>>>(v_f32, v_bf);

  attn_kernel<<<kB * kH, 256, 0, stream>>>(q_rope, k_rope, v_bf, ctx);

  gemm_xWT<true><<<dim3(kHID / 128, M / 128), blk, 0, stream>>>(ctx, wo, out, M, kHID, kHID);
}
